// GATLayer_61718680043880
// MI455X (gfx1250) — matrix operand order repaired and checked by reading, not compiled, not run on hardware
//
#include <hip/hip_runtime.h>
#include <math.h>

typedef __attribute__((ext_vector_type(16))) __bf16 v16bf;
typedef __attribute__((ext_vector_type(8)))  float  v8f;
typedef __attribute__((ext_vector_type(4)))  int    v4i;

#define B_    8
#define N_    2048
#define FIN_  256
#define FOUT_ 256
#define ALPHA 0.2f
#define NEG_BIG (-9000000000000000.0f)

#define AS1 __attribute__((address_space(1)))
#define AS3 __attribute__((address_space(3)))

#define ADJ_STRIDE 36      // 32 data ints + 4 pad: conflict-free reads, 16B-aligned rows
#define NBUF 8             // async pipeline depth (7 tiles in flight)

__device__ inline __bf16 to_bf16(float f) { return (__bf16)f; }   // native fptrunc (RNE)

__device__ inline unsigned pack2(float lo, float hi) {
    unsigned short a = __builtin_bit_cast(unsigned short, to_bf16(lo));
    unsigned short b = __builtin_bit_cast(unsigned short, to_bf16(hi));
    return (unsigned)a | ((unsigned)b << 16);
}

// --- async global->LDS copy (gfx1250 async path), with safe fallback -------
__device__ inline void copy16_async(const int4* gsrc, int* ldst) {
#if __has_builtin(__builtin_amdgcn_global_load_async_to_lds_b128)
    v4i* s = (v4i*)gsrc;                  // drop const + retype (generic)
    v4i* d = (v4i*)ldst;
    __builtin_amdgcn_global_load_async_to_lds_b128(
        (AS1 v4i*)s, (AS3 v4i*)d, 0, 0);  // addrspacecast generic->AS1/AS3
#else
    *reinterpret_cast<int4*>(ldst) = *gsrc;
#endif
}
template <int NWAIT>
__device__ inline void async_wait() {
#if __has_builtin(__builtin_amdgcn_s_wait_asynccnt)
    __builtin_amdgcn_s_wait_asynccnt(NWAIT);
#else
    asm volatile("s_wait_asynccnt %0" :: "i"(NWAIT) : "memory");
#endif
    asm volatile("" ::: "memory");
}

// ---------------------------------------------------------------------------
// Kernel 0: W (fp32, KxN row-major) -> Wt (bf16, [n][k], K-pairs contiguous)
// ---------------------------------------------------------------------------
__global__ __launch_bounds__(256)
void wconv_kernel(const float* __restrict__ W, unsigned short* __restrict__ Wt) {
    int t = blockIdx.x * 256 + threadIdx.x;        // 0..65535
    int n = t >> 8, k = t & 255;
    Wt[n * FIN_ + k] = __builtin_bit_cast(unsigned short, to_bf16(W[(size_t)k * FOUT_ + n]));
}

// ---------------------------------------------------------------------------
// Kernel 1: wh = h @ W (bf16 WMMA, fp32 acc); fused s1 = wh@a1, s2 = wh@a2.
// Stores wh^T (feature-major bf16) for the flash kernel's B operand.
// grid = B * (N/64), 128 threads (4 waves); each wave: 16 rows x 256.
// ---------------------------------------------------------------------------
__global__ __launch_bounds__(128)
void wh_kernel(const float* __restrict__ h, const unsigned short* __restrict__ Wt,
               const float* __restrict__ av, unsigned short* __restrict__ whT,
               float* __restrict__ s1, float* __restrict__ s2)
{
    const int lane = threadIdx.x & 31;
    const int wave = threadIdx.x >> 5;
    const int bat  = blockIdx.x >> 5;
    const int row0 = (blockIdx.x & 31) * 64 + wave * 16;
    const int m = lane & 15, g = lane >> 4;

    const float* hrow = h + ((size_t)bat * N_ + row0 + m) * FIN_;

    v8f acc[16] = {};

    #pragma unroll
    for (int ks = 0; ks < FIN_ / 32; ks++) {
        const int k0 = ks * 32;
        v16bf A;                                    // 16x32 A-layout tile of h
        #pragma unroll
        for (int v = 0; v < 8; v++) {
            int kb = k0 + 2 * v + g * 16;
            A[2 * v]     = to_bf16(hrow[kb]);
            A[2 * v + 1] = to_bf16(hrow[kb + 1]);
        }
        #pragma unroll
        for (int nt = 0; nt < 16; nt++) {
            v16bf Bv = *reinterpret_cast<const v16bf*>(
                Wt + (size_t)(nt * 16 + m) * FIN_ + k0 + g * 16);
            acc[nt] = __builtin_amdgcn_wmma_f32_16x16x32_bf16(
                false, A, false, Bv, (short)0, acc[nt], false, false);
        }
    }

    float s1p[8] = {}, s2p[8] = {};
    #pragma unroll
    for (int nt = 0; nt < 16; nt++) {
        const float a1v = av[nt * 16 + m];
        const float a2v = av[FOUT_ + nt * 16 + m];
        #pragma unroll
        for (int r = 0; r < 8; r++) {
            float val = acc[nt][r];                 // row = row0+r+8g, feat = nt*16+m
            s1p[r] += val * a1v;
            s2p[r] += val * a2v;
        }
        uint4 u4;
        u4.x = pack2(acc[nt][0], acc[nt][1]);
        u4.y = pack2(acc[nt][2], acc[nt][3]);
        u4.z = pack2(acc[nt][4], acc[nt][5]);
        u4.w = pack2(acc[nt][6], acc[nt][7]);
        size_t base = ((size_t)bat * FOUT_ + nt * 16 + m) * N_ + row0 + 8 * g;
        *reinterpret_cast<uint4*>(whT + base) = u4;
    }
    #pragma unroll
    for (int r = 0; r < 8; r++) {
        float v1 = s1p[r], v2 = s2p[r];
        for (int off = 1; off < 16; off <<= 1) {
            v1 += __shfl_xor(v1, off, 32);
            v2 += __shfl_xor(v2, off, 32);
        }
        if (m == 0) {
            s1[(size_t)bat * N_ + row0 + r + 8 * g] = v1;
            s2[(size_t)bat * N_ + row0 + r + 8 * g] = v2;
        }
    }
}

// ---------------------------------------------------------------------------
// Kernel 2: flash-style masked softmax(LeakyReLU(s1+s2)) @ wh.
// adj streamed via 7-deep async global->LDS pipeline (coalesced 16B/lane).
// grid = B * (N/64), 128 threads (4 waves); each wave one 16-row tile.
// ---------------------------------------------------------------------------
__global__ __launch_bounds__(128)
void attn_kernel(const int* __restrict__ adj, const unsigned short* __restrict__ whT,
                 const float* __restrict__ s1g, const float* __restrict__ s2g,
                 float* __restrict__ out)
{
    __shared__ float s2sh[N_];                         // 8 KB
    __shared__ int   adjsh[4][NBUF][16 * ADJ_STRIDE];  // 72 KB, per-wave ring

    const int lane = threadIdx.x & 31;
    const int wave = threadIdx.x >> 5;
    const int bat  = blockIdx.x >> 5;
    const int row0 = (blockIdx.x & 31) * 64 + wave * 16;
    const int m = lane & 15, g = lane >> 4;

    for (int i = threadIdx.x; i < N_; i += blockDim.x)
        s2sh[i] = s2g[(size_t)bat * N_ + i];
    __syncthreads();

    const float s1m = s1g[(size_t)bat * N_ + row0 + m];
    float mrun = -__builtin_inff();
    float lrun = 0.f;
    v8f accO[16] = {};

    const int lr = lane >> 1, ch = lane & 1;          // coalesced loader mapping
    const int4* adjg =
        reinterpret_cast<const int4*>(adj + (size_t)bat * N_ * N_ + (size_t)(row0 + lr) * N_);
    const unsigned short* whb = whT + (size_t)bat * FOUT_ * N_;

    // issue one 16x32 tile (4 x async b128 per lane) into ring buffer `buf`
    auto issue = [&](int ct, int buf) {
        const int4* p = adjg + ct * 8 + ch * 4;
        int* dst = &adjsh[wave][buf][lr * ADJ_STRIDE + ch * 16];
        #pragma unroll
        for (int q = 0; q < 4; q++) copy16_async(p + q, dst + q * 4);
    };

    constexpr int NT = N_ / 32;
    #pragma unroll
    for (int pf = 0; pf < NBUF - 1; pf++) issue(pf, pf);

    for (int ct = 0; ct < NT; ct++) {
        const int col0 = ct * 32;
        if (ct + NBUF - 1 < NT) { issue(ct + NBUF - 1, (ct + NBUF - 1) & (NBUF - 1));
                                  async_wait<4 * (NBUF - 1)>(); }  // oldest tile done
        else                    { async_wait<0>(); }
        const int* myAdj = &adjsh[wave][ct & (NBUF - 1)][0];

        // logits in WMMA A-layout order
        float p16[16];
        float tmax = -__builtin_inff();
        #pragma unroll
        for (int v = 0; v < 8; v++) {
            const int kb = 2 * v + g * 16;
            #pragma unroll
            for (int s = 0; s < 2; s++) {
                const int k = kb + s;
                const int aval = myAdj[m * ADJ_STRIDE + k];
                float e = s1m + s2sh[col0 + k];
                e = e > 0.f ? e : ALPHA * e;          // leaky_relu
                e = aval > 0 ? e : NEG_BIG;           // mask
                p16[2 * v + s] = e;
                tmax = fmaxf(tmax, e);
            }
        }
        tmax = fmaxf(tmax, __shfl_xor(tmax, 16, 32));
        const float mnew  = fmaxf(mrun, tmax);
        const float scale = __expf(mrun - mnew);      // 0 on first tile
        float rs = 0.f;
        v16bf A;
        #pragma unroll
        for (int i = 0; i < 16; i++) {
            float p = __expf(p16[i] - mnew);
            rs += p;
            A[i] = to_bf16(p);
        }
        rs += __shfl_xor(rs, 16, 32);
        lrun = lrun * scale + rs;
        mrun = mnew;

        // rescale only when the running max actually moved (rare after warmup)
        if (__any(scale != 1.f)) {
            float sc[8];
            #pragma unroll
            for (int r = 0; r < 8; r++) sc[r] = __shfl(scale, r + 8 * g, 32);
            #pragma unroll
            for (int nt = 0; nt < 16; nt++)
                #pragma unroll
                for (int r = 0; r < 8; r++) accO[nt][r] *= sc[r];
        }

        #pragma unroll
        for (int nt = 0; nt < 16; nt++) {
            v16bf Bv = *reinterpret_cast<const v16bf*>(
                whb + (size_t)(nt * 16 + m) * N_ + col0 + g * 16);
            accO[nt] = __builtin_amdgcn_wmma_f32_16x16x32_bf16(
                false, A, false, Bv, (short)0, accO[nt], false, false);
        }
    }

    const float linv = 1.f / lrun;
    float li[8];
    #pragma unroll
    for (int r = 0; r < 8; r++) li[r] = __shfl(linv, r + 8 * g, 32);
    #pragma unroll
    for (int nt = 0; nt < 16; nt++)
        #pragma unroll
        for (int r = 0; r < 8; r++)
            out[((size_t)bat * N_ + row0 + r + 8 * g) * FOUT_ + nt * 16 + m] =
                accO[nt][r] * li[r];
}

extern "C" void kernel_launch(void* const* d_in, const int* in_sizes, int n_in,
                              void* d_out, int out_size, void* d_ws, size_t ws_size,
                              hipStream_t stream) {
    const float* h   = (const float*)d_in[0];
    const int*   adj = (const int*)d_in[1];
    const float* W   = (const float*)d_in[2];
    const float* a   = (const float*)d_in[3];
    float* out = (float*)d_out;

    unsigned short* whT = (unsigned short*)d_ws;                  // 8 MB bf16 wh^T
    unsigned short* Wt  = whT + (size_t)B_ * FOUT_ * N_;          // 128 KB bf16 W^T
    float* s1 = (float*)(Wt + (size_t)FOUT_ * FIN_);              // 64 KB
    float* s2 = s1 + (size_t)B_ * N_;                             // 64 KB

    wconv_kernel<<<dim3(FOUT_ * FIN_ / 256), dim3(256), 0, stream>>>(W, Wt);
    wh_kernel<<<dim3(B_ * (N_ / 64)), dim3(128), 0, stream>>>(h, Wt, a, whT, s1, s2);
    attn_kernel<<<dim3(B_ * (N_ / 64)), dim3(128), 0, stream>>>(adj, whT, s1, s2, out);
}
